// NeuroVPR_DH_SNN_2637109920478
// MI455X (gfx1250) — compile-verified, matrix operand order checked by reading, not run on hardware
//
#include <hip/hip_runtime.h>
#include <hip/hip_bf16.h>
#include <math.h>

// ---------------- types ----------------
typedef __attribute__((ext_vector_type(16))) __bf16 v16bf;
typedef __attribute__((ext_vector_type(8)))  __bf16 v8bf;
typedef __attribute__((ext_vector_type(4)))  __bf16 v4bf;
typedef __attribute__((ext_vector_type(8)))  float  v8f;
typedef __attribute__((ext_vector_type(4)))  float  v4f;

#define NB      4
#define TSTEPS  100
#define BATCH   64
#define DIN     8192
#define SUBIN   2048
#define HD      256
#define SUBOUT  64
#define NCLASS  100

__device__ __forceinline__ v8f wmma_bf16(v16bf a, v16bf b, v8f c) {
  return __builtin_amdgcn_wmma_f32_16x16x32_bf16(false, a, false, b, (short)0, c, false, false);
}
// 3-pass split-precision accumulate: (ah+al)*(bh+bl) ~= ah*bh + ah*bl + al*bh
__device__ __forceinline__ v8f wmma_split(v16bf ah, v16bf al, v16bf bh, v16bf bl, v8f c) {
  c = wmma_bf16(ah, bh, c);
  c = wmma_bf16(ah, bl, c);
  c = wmma_bf16(al, bh, c);
  return c;
}
__device__ __forceinline__ void split1(float x, __bf16& h, __bf16& l) {
  h = (__bf16)x;
  l = (__bf16)(x - (float)h);
}

// low 32 bits of a generic pointer into LDS aperture == LDS byte offset
__device__ __forceinline__ unsigned lds_addr32(const void* p) {
  return (unsigned)(uintptr_t)p;
}
// async copy 16B global -> LDS (per lane), tracked by ASYNCcnt
__device__ __forceinline__ void async_g2l_b128(unsigned lds, const void* gptr, int ioff) {
  unsigned long long ga = (unsigned long long)(uintptr_t)gptr;
  asm volatile("global_load_async_to_lds_b128 %0, %1, off offset:%2"
               :: "v"(lds), "v"(ga), "i"(ioff) : "memory");
}
__device__ __forceinline__ void wait_asynccnt_8() {
  asm volatile("s_wait_asynccnt 8" ::: "memory");
}
__device__ __forceinline__ void wait_asynccnt_0() {
  asm volatile("s_wait_asynccnt 0" ::: "memory");
}

// ---------------- phase 0: weight split + gates ----------------
__global__ void k_split(const float* __restrict__ src, __bf16* __restrict__ h,
                        __bf16* __restrict__ l, int n) {
  int i = blockIdx.x * 256 + threadIdx.x;
  if (i < n) {
    float x = src[i];
    __bf16 hh, ll; split1(x, hh, ll);
    h[i] = hh; l[i] = ll;
  }
}

__global__ void k_sigmoid(const float* __restrict__ tn0, const float* __restrict__ tm0,
                          const float* __restrict__ tn1, const float* __restrict__ tm1,
                          float* __restrict__ b0, float* __restrict__ a0,
                          float* __restrict__ b1, float* __restrict__ a1) {
  int i = threadIdx.x;  // 256 threads, H = 256
  b0[i] = 1.f / (1.f + expf(-tn0[i]));
  a0[i] = 1.f / (1.f + expf(-tm0[i]));
  b1[i] = 1.f / (1.f + expf(-tn1[i]));
  a1[i] = 1.f / (1.f + expf(-tm1[i]));
}

// ---------------- phase 1: branch0[t,b,:] = blockdiag(x_t W0^T) ----------------
// grid = (400 M-tiles of 16 rows, 4 branches); 128 threads = 4 waves; wave = N-tile.
__global__ __launch_bounds__(128) void k_gemm0(
    const float* __restrict__ x,           // (B,T,8192)
    const __bf16* __restrict__ W0h,        // (4,64,2048)
    const __bf16* __restrict__ W0l,
    float* __restrict__ br0)               // (T,64,256)
{
  const int mt   = blockIdx.x;             // 0..399
  const int j    = blockIdx.y;             // branch
  const int tid  = threadIdx.x;
  const int wave = tid >> 5;
  const int lane = tid & 31;
  const int ln   = lane & 15;
  const int koff = (lane < 16) ? 0 : 8;

  const int t  = mt >> 2;                  // 16-row tile: t = (mt*16)/64
  const int b0 = (mt & 3) * 16;

  // A chunk staged as split-bf16; row stride 136 elems = 272B (16B aligned, 4-bank shift)
  __shared__ __bf16 Ah[16][136];
  __shared__ __bf16 Al[16][136];

  // B row pointers (one 16-col N-tile per wave)
  const int n = wave * 16 + ln;
  const __bf16* wrh = W0h + ((size_t)(j * SUBOUT + n)) * SUBIN;
  const __bf16* wrl = W0l + ((size_t)(j * SUBOUT + n)) * SUBIN;

  // cooperative A-loader mapping: 16 rows x 128 cols per chunk
  const int lm = tid >> 3;                 // 0..15
  const int lc = (tid & 7) * 16;           // 0..112
  const float* xrow = x + (((size_t)(b0 + lm)) * TSTEPS + t) * DIN + (size_t)j * SUBIN;

  v8f acc = {};
  for (int kc = 0; kc < SUBIN; kc += 128) {
    __syncthreads();
#pragma unroll
    for (int i = 0; i < 4; ++i) {
      v4f f = *(const v4f*)(xrow + kc + lc + i * 4);
      v4bf h4, l4;
#pragma unroll
      for (int e = 0; e < 4; ++e) { __bf16 hh, ll; split1(f[e], hh, ll); h4[e] = hh; l4[e] = ll; }
      *(v4bf*)&Ah[lm][lc + i * 4] = h4;
      *(v4bf*)&Al[lm][lc + i * 4] = l4;
    }
    __syncthreads();
#pragma unroll
    for (int ks = 0; ks < 128; ks += 32) {
      v16bf ah, al, bh, bl;
      ((v8bf*)&ah)[0] = *(const v8bf*)&Ah[ln][ks + koff];
      ((v8bf*)&ah)[1] = *(const v8bf*)&Ah[ln][ks + 16 + koff];
      ((v8bf*)&al)[0] = *(const v8bf*)&Al[ln][ks + koff];
      ((v8bf*)&al)[1] = *(const v8bf*)&Al[ln][ks + 16 + koff];
      const int kg = kc + ks;
      ((v8bf*)&bh)[0] = *(const v8bf*)(wrh + kg + koff);
      ((v8bf*)&bh)[1] = *(const v8bf*)(wrh + kg + 16 + koff);
      ((v8bf*)&bl)[0] = *(const v8bf*)(wrl + kg + koff);
      ((v8bf*)&bl)[1] = *(const v8bf*)(wrl + kg + 16 + koff);
      acc = wmma_split(ah, al, bh, bl, acc);
    }
  }
  // C layout: elem q -> (M = q + 8*(lane>=16), N = lane&15)
  const int hi = (lane >= 16) ? 8 : 0;
  float* orow = br0 + (size_t)t * BATCH * HD;
#pragma unroll
  for (int q = 0; q < 8; ++q)
    orow[(size_t)(b0 + q + hi) * HD + j * SUBOUT + wave * 16 + ln] = acc[q];
}

// ---------------- phase 2: sequential recurrence + readout ----------------
// 4 blocks (16 batch rows each) x 128 threads; wave j owns feature columns [64j, 64j+64).
// branch0[t] tiles are double-buffered into LDS with global_load_async_to_lds_b128
// one step ahead, so the serial chain never waits on VMEM latency.
__global__ __launch_bounds__(128) void k_recur(
    const float* __restrict__ br0,          // (T,64,256)
    const __bf16* __restrict__ W1h,         // (4,64,64)
    const __bf16* __restrict__ W1l,
    const float* __restrict__ beta0, const float* __restrict__ alpha0,
    const float* __restrict__ beta1, const float* __restrict__ alpha1,
    const __bf16* __restrict__ Wouth,       // (100,256)
    const __bf16* __restrict__ Woutl,
    const float* __restrict__ bout,         // (100)
    const float* __restrict__ v00,          // (64,256)
    const float* __restrict__ v01,          // (64,256)
    float* __restrict__ out)                // (64,100)
{
  const int b0   = blockIdx.x * 16;
  const int tid  = threadIdx.x;
  const int j    = tid >> 5;                // wave == branch
  const int lane = tid & 31;
  const int ln   = lane & 15;
  const int hi   = (lane >= 16) ? 8 : 0;
  const int koff = (lane < 16) ? 0 : 8;

  __shared__ float s0s[NB][16][68];         // per-wave s0 slice (272B rows: aligned, bank-shifted)
  __shared__ union {
    float br[2][16][HD];                    // double-buffered branch0 tile (32KB)
    float ssum[16][264];                    // readout buffer (reused after the t-loop)
  } sh;

  // async prefetch mapping: thread covers 32 consecutive floats of one row
  const int prow = tid >> 3;                // 0..15
  const int pcol = (tid & 7) * 32;          // 0..224
  const unsigned ldsb[2] = { lds_addr32(&sh.br[0][prow][pcol]),
                             lds_addr32(&sh.br[1][prow][pcol]) };

  // per-tile gate constants (column h = 64j + 16*nt + ln, independent of q)
  float be0[4], a0c[4], be1[4], a1c[4];
#pragma unroll
  for (int nt = 0; nt < 4; ++nt) {
    int h = j * SUBOUT + nt * 16 + ln;
    be0[nt] = beta0[h];  a0c[nt] = alpha0[h];
    be1[nt] = beta1[h];  a1c[nt] = alpha1[h];
  }

  // stage W1 B-fragments in registers once (branch j, 4 N-tiles x 2 K-steps, hi/lo)
  v16bf wbh[4][2], wbl[4][2];
#pragma unroll
  for (int nt = 0; nt < 4; ++nt) {
    const __bf16* wh = W1h + ((size_t)(j * SUBOUT + nt * 16 + ln)) * SUBOUT;
    const __bf16* wl = W1l + ((size_t)(j * SUBOUT + nt * 16 + ln)) * SUBOUT;
#pragma unroll
    for (int ks = 0; ks < 2; ++ks) {
      ((v8bf*)&wbh[nt][ks])[0] = *(const v8bf*)(wh + ks * 32 + koff);
      ((v8bf*)&wbh[nt][ks])[1] = *(const v8bf*)(wh + ks * 32 + 16 + koff);
      ((v8bf*)&wbl[nt][ks])[0] = *(const v8bf*)(wl + ks * 32 + koff);
      ((v8bf*)&wbl[nt][ks])[1] = *(const v8bf*)(wl + ks * 32 + 16 + koff);
    }
  }

  // recurrent state, WMMA C-layout fragments
  v8f d0f[4] = {}, s0f[4] = {}, d1f[4] = {}, s1f[4] = {}, s1sm[4] = {};
  v8f v0f[4], v1f[4];
#pragma unroll
  for (int nt = 0; nt < 4; ++nt)
#pragma unroll
    for (int q = 0; q < 8; ++q) {
      int h = j * SUBOUT + nt * 16 + ln;
      v0f[nt][q] = v00[(size_t)(b0 + q + hi) * HD + h];
      v1f[nt][q] = v01[(size_t)(b0 + q + hi) * HD + h];
    }

  // prologue: prefetch t=0 tile into buffer 0 (8 async b128 per wave)
  {
    const float* g = br0 + ((size_t)0 * BATCH + b0 + prow) * HD + pcol;
#pragma unroll
    for (int k = 0; k < 8; ++k) async_g2l_b128(ldsb[0], g, k * 16);
  }

  for (int t = 0; t < TSTEPS; ++t) {
    const int cur = t & 1;
    // prefetch t+1 into the other buffer while we compute on `cur`
    if (t + 1 < TSTEPS) {
      const float* g = br0 + ((size_t)(t + 1) * BATCH + b0 + prow) * HD + pcol;
#pragma unroll
      for (int k = 0; k < 8; ++k) async_g2l_b128(ldsb[cur ^ 1], g, k * 16);
      wait_asynccnt_8();                    // oldest 8 (tile t) have landed
    } else {
      wait_asynccnt_0();
    }
    __syncthreads();                        // all waves' tile-t data visible

    // ---- layer 0: elementwise update from LDS tile, write s0 slice to LDS ----
#pragma unroll
    for (int nt = 0; nt < 4; ++nt) {
#pragma unroll
      for (int q = 0; q < 8; ++q) {
        int M = q + hi;
        float br = sh.br[cur][M][j * SUBOUT + nt * 16 + ln];
        float d = be0[nt] * d0f[nt][q] + (1.f - be0[nt]) * br;
        d0f[nt][q] = d;
        float v = a0c[nt] * v0f[nt][q] + (1.f - a0c[nt]) * d - s0f[nt][q];
        v0f[nt][q] = v;
        float u = v - 1.f;
        float s = (u > 0.f) ? 1.f : 0.5f * tanhf(2.f * u);
        s0f[nt][q] = s;
        s0s[j][M][nt * 16 + ln] = s;
      }
    }
    __syncthreads();  // ordering fence for LDS re-layout (slice is wave-private)

    // ---- layer 1: 16x64 @ 64x64 via split-bf16 WMMA, then elementwise ----
#pragma unroll
    for (int nt = 0; nt < 4; ++nt) {
      v8f acc = {};
#pragma unroll
      for (int ks = 0; ks < 2; ++ks) {
        float fl[16];
        *(v4f*)&fl[0]  = *(const v4f*)&s0s[j][ln][ks * 32 + koff];
        *(v4f*)&fl[4]  = *(const v4f*)&s0s[j][ln][ks * 32 + koff + 4];
        *(v4f*)&fl[8]  = *(const v4f*)&s0s[j][ln][ks * 32 + 16 + koff];
        *(v4f*)&fl[12] = *(const v4f*)&s0s[j][ln][ks * 32 + 16 + koff + 4];
        v16bf ah, al;
#pragma unroll
        for (int e = 0; e < 16; ++e) { __bf16 hh, ll; split1(fl[e], hh, ll); ah[e] = hh; al[e] = ll; }
        acc = wmma_split(ah, al, wbh[nt][ks], wbl[nt][ks], acc);
      }
#pragma unroll
      for (int q = 0; q < 8; ++q) {
        float d = be1[nt] * d1f[nt][q] + (1.f - be1[nt]) * acc[q];
        d1f[nt][q] = d;
        float v = a1c[nt] * v1f[nt][q] + (1.f - a1c[nt]) * d - s1f[nt][q];
        v1f[nt][q] = v;
        float u = v - 1.f;
        float s = (u > 0.f) ? 1.f : 0.5f * tanhf(2.f * u);
        s1f[nt][q] = s;
        if (t >= TSTEPS - 3) s1sm[nt][q] += s;
      }
    }
    __syncthreads();  // reads of br[cur] done before next step's async refill
  }

  // ---- readout: out = (sum s1 / 3) @ Wout^T + bout ----
#pragma unroll
  for (int nt = 0; nt < 4; ++nt)
#pragma unroll
    for (int q = 0; q < 8; ++q)
      sh.ssum[q + hi][j * SUBOUT + nt * 16 + ln] = s1sm[nt][q] * (1.f / 3.f);
  __syncthreads();

  for (int n = j; n < 7; n += 4) {          // 7 N-tiles cover 112 >= 100 cols
    int col = n * 16 + ln;
    int wrow = (col < NCLASS) ? col : (NCLASS - 1);
    const __bf16* whp = Wouth + (size_t)wrow * HD;
    const __bf16* wlp = Woutl + (size_t)wrow * HD;
    v8f acc = {};
#pragma unroll
    for (int ks = 0; ks < 8; ++ks) {
      float fl[16];
      *(v4f*)&fl[0]  = *(const v4f*)&sh.ssum[ln][ks * 32 + koff];
      *(v4f*)&fl[4]  = *(const v4f*)&sh.ssum[ln][ks * 32 + koff + 4];
      *(v4f*)&fl[8]  = *(const v4f*)&sh.ssum[ln][ks * 32 + 16 + koff];
      *(v4f*)&fl[12] = *(const v4f*)&sh.ssum[ln][ks * 32 + 16 + koff + 4];
      v16bf ah, al, bh, bl;
#pragma unroll
      for (int e = 0; e < 16; ++e) { __bf16 hh, ll; split1(fl[e], hh, ll); ah[e] = hh; al[e] = ll; }
      ((v8bf*)&bh)[0] = *(const v8bf*)(whp + ks * 32 + koff);
      ((v8bf*)&bh)[1] = *(const v8bf*)(whp + ks * 32 + 16 + koff);
      ((v8bf*)&bl)[0] = *(const v8bf*)(wlp + ks * 32 + koff);
      ((v8bf*)&bl)[1] = *(const v8bf*)(wlp + ks * 32 + 16 + koff);
      acc = wmma_split(ah, al, bh, bl, acc);
    }
    if (col < NCLASS) {
      float bb = bout[col];
#pragma unroll
      for (int q = 0; q < 8; ++q)
        out[(size_t)(b0 + q + hi) * NCLASS + col] = acc[q] + bb;
    }
  }
}

// ---------------- launcher ----------------
extern "C" void kernel_launch(void* const* d_in, const int* in_sizes, int n_in,
                              void* d_out, int out_size, void* d_ws, size_t ws_size,
                              hipStream_t stream) {
  (void)in_sizes; (void)n_in; (void)out_size; (void)ws_size;
  const float* x      = (const float*)d_in[0];   // (64,100,2,64,64)
  const float* W0     = (const float*)d_in[1];   // (4,64,2048)
  const float* tau_n0 = (const float*)d_in[2];   // (4,64)
  const float* tau_m0 = (const float*)d_in[3];   // (256)
  const float* W1     = (const float*)d_in[4];   // (4,64,64)
  const float* tau_n1 = (const float*)d_in[5];   // (4,64)
  const float* tau_m1 = (const float*)d_in[6];   // (256)
  const float* Wout   = (const float*)d_in[7];   // (100,256)
  const float* bout   = (const float*)d_in[8];   // (100)
  const float* v00    = (const float*)d_in[9];   // (64,256)
  const float* v01    = (const float*)d_in[10];  // (64,256)
  float* out = (float*)d_out;

  // workspace carving (256B aligned)
  size_t off = 0;
  auto carve = [&](size_t bytes) -> void* {
    void* p = (char*)d_ws + off;
    off += (bytes + 255) & ~(size_t)255;
    return p;
  };
  const size_t nW0 = (size_t)NB * SUBOUT * SUBIN;   // 524288
  const size_t nW1 = (size_t)NB * SUBOUT * SUBOUT;  // 16384
  const size_t nWo = (size_t)NCLASS * HD;           // 25600
  float*  br0    = (float*)carve((size_t)TSTEPS * BATCH * HD * sizeof(float));
  __bf16* W0h    = (__bf16*)carve(nW0 * 2);
  __bf16* W0l    = (__bf16*)carve(nW0 * 2);
  __bf16* W1h    = (__bf16*)carve(nW1 * 2);
  __bf16* W1l    = (__bf16*)carve(nW1 * 2);
  __bf16* Wouth  = (__bf16*)carve(nWo * 2);
  __bf16* Woutl  = (__bf16*)carve(nWo * 2);
  float*  beta0  = (float*)carve(HD * sizeof(float));
  float*  alpha0 = (float*)carve(HD * sizeof(float));
  float*  beta1  = (float*)carve(HD * sizeof(float));
  float*  alpha1 = (float*)carve(HD * sizeof(float));

  k_split<<<(int)((nW0 + 255) / 256), 256, 0, stream>>>(W0, W0h, W0l, (int)nW0);
  k_split<<<(int)((nW1 + 255) / 256), 256, 0, stream>>>(W1, W1h, W1l, (int)nW1);
  k_split<<<(int)((nWo + 255) / 256), 256, 0, stream>>>(Wout, Wouth, Woutl, (int)nWo);
  k_sigmoid<<<1, 256, 0, stream>>>(tau_n0, tau_m0, tau_n1, tau_m1,
                                   beta0, alpha0, beta1, alpha1);

  k_gemm0<<<dim3((TSTEPS * BATCH) / 16, NB), 128, 0, stream>>>(x, W0h, W0l, br0);

  k_recur<<<BATCH / 16, 128, 0, stream>>>(br0, W1h, W1l,
                                          beta0, alpha0, beta1, alpha1,
                                          Wouth, Woutl, bout, v00, v01, out);
}
